// DecoderFusionBlock_51780125720658
// MI455X (gfx1250) — matrix operand order repaired and checked by reading, not compiled, not run on hardware
//
#include <hip/hip_runtime.h>
#include <hip/hip_bf16.h>

typedef __attribute__((ext_vector_type(16))) _Float16 v16h;
typedef __attribute__((ext_vector_type(8)))  float    v8f;

namespace {
constexpr int B_   = 4;
constexpr int H_   = 48;
constexpr int W_   = 48;
constexpr int DIN  = 192;
constexpr int NST  = 16;
constexpr int RNK  = 6;
constexpr int L_   = H_ * W_;   // 2304
constexpr int NTOK = B_ * L_;   // 9216
}

// ---------------------------------------------------------------------------
// WMMA helpers (CDNA5 v_wmma_f32_16x16x32_f16, wave32)
// ---------------------------------------------------------------------------
__device__ __forceinline__ v8f wmma32f16(v16h a, v16h b, v8f c) {
  return __builtin_amdgcn_wmma_f32_16x16x32_f16(
      /*neg_a=*/false, a, /*neg_b=*/false, b,
      /*c_mod=*/(short)0, c, /*reuse_a=*/false, /*reuse_b=*/false);
}

// Gather A fragment (16x32 f16) from an LDS tile, row-major with stride lda.
// Contiguous-k layout used here (same as prepack): lane m=lane&15; VGPR v: kbase = (lane>>4)*16 + v*2
__device__ __forceinline__ v16h fragA(const _Float16* As, int lda, int lane) {
  v16h a;
  int m  = lane & 15;
  int hi = lane >> 4;
#pragma unroll
  for (int v = 0; v < 8; ++v) {
    int kb = (hi << 4) + (v << 1);
    const _Float16* p = As + m * lda + kb;
    a[2 * v]     = p[0];
    a[2 * v + 1] = p[1];
  }
  return a;
}

// ---------------------------------------------------------------------------
// CDNA5 async global->LDS copy (VGPR-bypassing, tracked by ASYNCcnt).
// Low 32 bits of a generic pointer to a __shared__ object are the AS3 offset
// (flat aperture rule).
// ---------------------------------------------------------------------------
__device__ __forceinline__ void async_lds_b128(void* lds_dst, const void* gsrc) {
  unsigned int l = (unsigned int)(size_t)lds_dst;
  unsigned long long g = (unsigned long long)(size_t)gsrc;
  asm volatile("global_load_async_to_lds_b128 %0, %1, off"
               :: "v"(l), "v"(g) : "memory");
}
__device__ __forceinline__ void wait_async0() {
  asm volatile("s_wait_asynccnt 0" ::: "memory");
}

// ---------------------------------------------------------------------------
// Weight prepack: f32 -> f16 fragments. Frag element j of lane corresponds to
// k_local = (lane>>4)*16 + j, n = nt*16 + (lane&15).  dst frag = nt*KT + kt.
// mode 0: src row-major [K][N];  mode 1: src transposed [N][K];
// mode 2: x_proj special: n -> (kdir=n/48, c=n%48), valid c<38, src[(kdir*38+c)*192 + k]
// ---------------------------------------------------------------------------
__global__ __launch_bounds__(128) void k_prepack(const float* __restrict__ src,
                                                 _Float16* __restrict__ dst,
                                                 int K, int N, int KT, int NT, int mode) {
  int t = blockIdx.x * 128 + threadIdx.x;
  int total = NT * KT * 32;
  if (t >= total) return;
  int lane = t & 31;
  int f    = t >> 5;          // frag index = nt*KT + kt
  int kt   = f % KT;
  int nt   = f / KT;
  int n  = nt * 16 + (lane & 15);
  int hi = lane >> 4;
  v16h o;
#pragma unroll
  for (int j = 0; j < 16; ++j) {
    int kk  = kt * 32 + hi * 16 + j;
    float v = 0.f;
    if (kk < K && n < N) {
      if (mode == 0)      v = src[(size_t)kk * N + n];
      else if (mode == 1) v = src[(size_t)n * K + kk];
      else {              // x_proj_W [4][38][192]
        int kd = n / 48, c = n % 48;
        if (c < 38) v = src[((size_t)kd * 38 + c) * 192 + kk];
      }
    }
    o[j] = (_Float16)v;
  }
  ((v16h*)dst)[(size_t)f * 32 + lane] = o;
}

// ---------------------------------------------------------------------------
// Kernel A: x = x_cat@proj_W+b ; LN1 ; xz = xn@in_proj_W+b ; split xm/z
// One block = 16 tokens, 4 waves.  (A source is f32 -> VALU convert staging.)
// ---------------------------------------------------------------------------
__global__ __launch_bounds__(128) void k_proj_ln_inproj(
    const float* __restrict__ xcat, const float* __restrict__ proj_b,
    const float* __restrict__ ln1w, const float* __restrict__ ln1b,
    const float* __restrict__ inpb,
    const _Float16* __restrict__ Wp1, const _Float16* __restrict__ Wp2,
    float* __restrict__ x_res, float* __restrict__ xm_out, float* __restrict__ z_out) {
  __shared__ __align__(16) _Float16 As[16][200];
  __shared__ float    Cx[16][96];
  __shared__ __align__(16) _Float16 An[16][104];
  int tok0 = blockIdx.x * 16;
  int tid = threadIdx.x, lane = tid & 31, wave = tid >> 5;

  for (int i = tid; i < 16 * 192; i += 128) {
    int m = i / 192, c = i % 192;
    As[m][c] = (_Float16)xcat[(size_t)(tok0 + m) * 192 + c];
  }
  __syncthreads();

  const v16h* B1 = (const v16h*)Wp1;   // KT=6, NT=6
  for (int nt = wave; nt < 6; nt += 4) {
    v8f acc = {};
#pragma unroll
    for (int kt = 0; kt < 6; ++kt) {
      v16h a = fragA(&As[0][0] + kt * 32, 200, lane);
      v16h b = B1[(size_t)(nt * 6 + kt) * 32 + lane];
      acc = wmma32f16(a, b, acc);
    }
    int n = nt * 16 + (lane & 15);
    float bias = proj_b[n];
    int hi = lane >> 4;
#pragma unroll
    for (int v = 0; v < 8; ++v) Cx[hi * 8 + v][n] = acc[v] + bias;
  }
  __syncthreads();

  for (int i = tid; i < 16 * 96; i += 128) {
    int m = i / 96, c = i % 96;
    x_res[(size_t)(tok0 + m) * 96 + c] = Cx[m][c];
  }
  if (tid < 16) {
    float mean = 0.f, var = 0.f;
    for (int c = 0; c < 96; ++c) mean += Cx[tid][c];
    mean *= (1.f / 96.f);
    for (int c = 0; c < 96; ++c) { float d = Cx[tid][c] - mean; var += d * d; }
    var *= (1.f / 96.f);
    float r = rsqrtf(var + 1e-5f);
    for (int c = 0; c < 96; ++c)
      An[tid][c] = (_Float16)(((Cx[tid][c] - mean) * r) * ln1w[c] + ln1b[c]);
  }
  __syncthreads();

  const v16h* B2 = (const v16h*)Wp2;   // KT=3, NT=24
  for (int nt = wave; nt < 24; nt += 4) {
    v8f acc = {};
#pragma unroll
    for (int kt = 0; kt < 3; ++kt) {
      v16h a = fragA(&An[0][0] + kt * 32, 104, lane);
      v16h b = B2[(size_t)(nt * 3 + kt) * 32 + lane];
      acc = wmma32f16(a, b, acc);
    }
    int n = nt * 16 + (lane & 15);
    float bias = inpb[n];
    int hi = lane >> 4;
#pragma unroll
    for (int v = 0; v < 8; ++v) {
      int tok = tok0 + hi * 8 + v;
      float val = acc[v] + bias;
      if (n < 192) xm_out[(size_t)tok * 192 + n] = val;
      else         z_out[(size_t)tok * 192 + (n - 192)] = val;
    }
  }
}

// ---------------------------------------------------------------------------
// Depthwise 3x3 (SAME) + bias + SiLU :  xm (NHWC f32) -> u (f16)
// ---------------------------------------------------------------------------
__global__ __launch_bounds__(256) void k_dwconv_silu(
    const float* __restrict__ xm, const float* __restrict__ cw,
    const float* __restrict__ cb, _Float16* __restrict__ u) {
  int idx = blockIdx.x * 256 + threadIdx.x;
  if (idx >= NTOK * DIN) return;
  int d = idx % DIN, tok = idx / DIN;
  int b = tok / L_, pos = tok % L_, h = pos / W_, w = pos % W_;
  float s = 0.f;
#pragma unroll
  for (int i = 0; i < 3; ++i)
#pragma unroll
    for (int j = 0; j < 3; ++j) {
      int hh = h + i - 1, ww = w + j - 1;
      if (hh >= 0 && hh < H_ && ww >= 0 && ww < W_)
        s += xm[((size_t)b * L_ + hh * W_ + ww) * DIN + d] * cw[d * 9 + i * 3 + j];
    }
  s += cb[d];
  float sig = 1.f / (1.f + __expf(-s));
  u[idx] = (_Float16)(s * sig);
}

__device__ __forceinline__ int seq_of_pos(int k, int pos) {
  int h = pos / W_, w = pos % W_;
  if (k == 0) return pos;
  if (k == 1) return w * H_ + h;
  if (k == 2) return L_ - 1 - pos;
  return L_ - 1 - (w * H_ + h);
}

// ---------------------------------------------------------------------------
// x_dbl = u @ Wproj (4 dirs, padded N=192), then delta=softplus(dt), scatter
// Bs/Cs/delta into per-direction sequence order.  A-tile staged via async LDS.
// ---------------------------------------------------------------------------
__global__ __launch_bounds__(128) void k_xproj(
    const _Float16* __restrict__ u, const _Float16* __restrict__ Wp3,
    const float* __restrict__ dtW, const float* __restrict__ dtb,
    float* __restrict__ delta_g, float* __restrict__ Bs_g, float* __restrict__ Cs_g) {
  __shared__ __align__(16) _Float16 Au[16][200];
  __shared__ float    Xd[16][192];
  int tok0 = blockIdx.x * 16;
  int tid = threadIdx.x, lane = tid & 31, wave = tid >> 5;

  // 16 rows x 192 halfs = 24 x 16B chunks per row, async global->LDS
  for (int c = tid; c < 16 * 24; c += 128) {
    int m = c / 24, q = c % 24;
    async_lds_b128(&Au[m][q * 8], u + (size_t)(tok0 + m) * 192 + q * 8);
  }
  wait_async0();
  __syncthreads();

  const v16h* Bw = (const v16h*)Wp3;   // KT=6, NT=12
  for (int nt = wave; nt < 12; nt += 4) {
    v8f acc = {};
#pragma unroll
    for (int kt = 0; kt < 6; ++kt) {
      v16h a = fragA(&Au[0][0] + kt * 32, 200, lane);
      v16h b = Bw[(size_t)(nt * 6 + kt) * 32 + lane];
      acc = wmma32f16(a, b, acc);
    }
    int n = nt * 16 + (lane & 15);
    int hi = lane >> 4;
#pragma unroll
    for (int v = 0; v < 8; ++v) Xd[hi * 8 + v][n] = acc[v];
  }
  __syncthreads();

  // Bs / Cs scatter
  for (int i = tid; i < 16 * 4 * 16; i += 128) {
    int n = i & 15, k = (i >> 4) & 3, m = i >> 6;
    int tok = tok0 + m, b = tok / L_, pos = tok % L_;
    long base = (long)(b * 4 + k) * L_ + seq_of_pos(k, pos);
    Bs_g[base * 16 + n] = Xd[m][k * 48 + RNK + n];
    Cs_g[base * 16 + n] = Xd[m][k * 48 + RNK + NST + n];
  }
  // delta = softplus(rank @ dt_W + dt_b)
  for (int i = tid; i < 16 * 4 * 192; i += 128) {
    int d = i % 192, k = (i / 192) & 3, m = i / 768;
    int tok = tok0 + m, b = tok / L_, pos = tok % L_;
    long base = (long)(b * 4 + k) * L_ + seq_of_pos(k, pos);
    float v = dtb[k * DIN + d];
#pragma unroll
    for (int r = 0; r < RNK; ++r)
      v += Xd[m][k * 48 + r] * dtW[((size_t)k * DIN + d) * RNK + r];
    float sp = (v > 20.f) ? v : log1pf(__expf(v));
    delta_g[base * 192 + d] = sp;
  }
}

// ---------------------------------------------------------------------------
// Selective scan: 96 wave32 waves; each lane owns one (b,k,d), 16 states in
// VGPRs.  ys written back in spatial (pos) order so merge is elementwise.
// ---------------------------------------------------------------------------
__global__ __launch_bounds__(32) void k_scan(
    const float* __restrict__ delta_g, const float* __restrict__ Bs_g,
    const float* __restrict__ Cs_g, const _Float16* __restrict__ u,
    const float* __restrict__ A_logs, const float* __restrict__ Ds,
    float* __restrict__ ys) {
  int bk = blockIdx.x / 6, dch = blockIdx.x % 6;
  int b = bk >> 2, k = bk & 3;
  int d = dch * 32 + threadIdx.x;
  float a[NST], h[NST];
#pragma unroll
  for (int n = 0; n < NST; ++n) {
    a[n] = -__expf(A_logs[((size_t)k * DIN + d) * NST + n]);
    h[n] = 0.f;
  }
  float Dval = Ds[k * DIN + d];
  const long baseL = (long)bk * L_;
  for (int l = 0; l < L_; ++l) {
    long base = baseL + l;
    float dl = delta_g[base * 192 + d];
    int pos;
    if (k == 0)      pos = l;
    else if (k == 1) pos = (l % H_) * W_ + l / H_;
    else if (k == 2) pos = L_ - 1 - l;
    else { int m2 = L_ - 1 - l; pos = (m2 % H_) * W_ + m2 / H_; }
    float uv = (float)u[((size_t)b * L_ + pos) * 192 + d];
    float du = dl * uv;
    float Bv[NST], Cv[NST];
    const float4* Bp = (const float4*)(Bs_g + base * 16);
    const float4* Cp = (const float4*)(Cs_g + base * 16);
#pragma unroll
    for (int q = 0; q < 4; ++q) {
      float4 t4 = Bp[q];
      Bv[4 * q] = t4.x; Bv[4 * q + 1] = t4.y; Bv[4 * q + 2] = t4.z; Bv[4 * q + 3] = t4.w;
      float4 c4 = Cp[q];
      Cv[4 * q] = c4.x; Cv[4 * q + 1] = c4.y; Cv[4 * q + 2] = c4.z; Cv[4 * q + 3] = c4.w;
    }
    if (l + 1 < L_) __builtin_prefetch(delta_g + (base + 1) * 192 + d, 0, 1);
    float y = 0.f;
#pragma unroll
    for (int n = 0; n < NST; ++n) {
      float da = __expf(dl * a[n]);
      h[n] = fmaf(da, h[n], du * Bv[n]);
      y = fmaf(h[n], Cv[n], y);
    }
    ys[((size_t)k * NTOK + (size_t)b * L_ + pos) * 192 + d] = y + uv * Dval;
  }
}

// ---------------------------------------------------------------------------
// Merge 4 directions + out_norm LN + silu(z) gate -> yg (f16)
// ---------------------------------------------------------------------------
__global__ __launch_bounds__(192) void k_merge_ln_gate(
    const float* __restrict__ ys, const float* __restrict__ z_g,
    const float* __restrict__ onw, const float* __restrict__ onb,
    _Float16* __restrict__ yg) {
  __shared__ float ba[256], bb[256];
  int tok = blockIdx.x, d = threadIdx.x;
  float y = 0.f;
#pragma unroll
  for (int k = 0; k < 4; ++k) y += ys[((size_t)k * NTOK + tok) * 192 + d];
  ba[d] = y; bb[d] = y * y;
  if (d < 64) { ba[192 + d] = 0.f; bb[192 + d] = 0.f; }
  __syncthreads();
  for (int s = 128; s > 0; s >>= 1) {
    if (d < s) { ba[d] += ba[d + s]; bb[d] += bb[d + s]; }
    __syncthreads();
  }
  float mean = ba[0] * (1.f / 192.f);
  float var  = bb[0] * (1.f / 192.f) - mean * mean;
  float r    = rsqrtf(var + 1e-5f);
  float yn   = (y - mean) * r * onw[d] + onb[d];
  float zv   = z_g[(size_t)tok * 192 + d];
  float sz   = zv / (1.f + __expf(-zv));
  yg[(size_t)tok * 192 + d] = (_Float16)(yn * sz);
}

// ---------------------------------------------------------------------------
// x2 = x + yg @ out_proj  (K=192, N=96)   -- async LDS A staging
// ---------------------------------------------------------------------------
__global__ __launch_bounds__(128) void k_outproj(
    const _Float16* __restrict__ yg, const _Float16* __restrict__ Wp4,
    const float* __restrict__ x_res, float* __restrict__ x2,
    _Float16* __restrict__ x2h) {
  __shared__ __align__(16) _Float16 Ay[16][200];
  int tok0 = blockIdx.x * 16;
  int tid = threadIdx.x, lane = tid & 31, wave = tid >> 5;
  for (int c = tid; c < 16 * 24; c += 128) {
    int m = c / 24, q = c % 24;
    async_lds_b128(&Ay[m][q * 8], yg + (size_t)(tok0 + m) * 192 + q * 8);
  }
  wait_async0();
  __syncthreads();
  const v16h* Bw = (const v16h*)Wp4;   // KT=6, NT=6
  for (int nt = wave; nt < 6; nt += 4) {
    v8f acc = {};
#pragma unroll
    for (int kt = 0; kt < 6; ++kt) {
      v16h a = fragA(&Ay[0][0] + kt * 32, 200, lane);
      v16h b = Bw[(size_t)(nt * 6 + kt) * 32 + lane];
      acc = wmma32f16(a, b, acc);
    }
    int n = nt * 16 + (lane & 15);
    int hi = lane >> 4;
#pragma unroll
    for (int v = 0; v < 8; ++v) {
      int tok = tok0 + hi * 8 + v;
      float val = acc[v] + x_res[(size_t)tok * 96 + n];
      x2[(size_t)tok * 96 + n]  = val;
      x2h[(size_t)tok * 96 + n] = (_Float16)val;
    }
  }
}

// ---------------------------------------------------------------------------
// pw1 (96->192) + bn1 + exact GELU   -- async LDS A staging
// ---------------------------------------------------------------------------
__global__ __launch_bounds__(128) void k_pw1(
    const _Float16* __restrict__ x2h, const _Float16* __restrict__ Wp5,
    const float* __restrict__ g1, const float* __restrict__ b1,
    _Float16* __restrict__ t1) {
  __shared__ __align__(16) _Float16 Ax[16][104];
  int tok0 = blockIdx.x * 16;
  int tid = threadIdx.x, lane = tid & 31, wave = tid >> 5;
  for (int c = tid; c < 16 * 12; c += 128) {
    int m = c / 12, q = c % 12;
    async_lds_b128(&Ax[m][q * 8], x2h + (size_t)(tok0 + m) * 96 + q * 8);
  }
  wait_async0();
  __syncthreads();
  const v16h* Bw = (const v16h*)Wp5;   // KT=3, NT=12
  for (int nt = wave; nt < 12; nt += 4) {
    v8f acc = {};
#pragma unroll
    for (int kt = 0; kt < 3; ++kt) {
      v16h a = fragA(&Ax[0][0] + kt * 32, 104, lane);
      v16h b = Bw[(size_t)(nt * 3 + kt) * 32 + lane];
      acc = wmma32f16(a, b, acc);
    }
    int n = nt * 16 + (lane & 15);
    float gg = g1[n], bb = b1[n];
    int hi = lane >> 4;
#pragma unroll
    for (int v = 0; v < 8; ++v) {
      int tok = tok0 + hi * 8 + v;
      float val = acc[v] * gg + bb;
      float ge = 0.5f * val * (1.f + erff(val * 0.70710678f));
      t1[(size_t)tok * 192 + n] = (_Float16)ge;
    }
  }
}

// ---------------------------------------------------------------------------
// Depthwise 3x3 + bn2 + GELU
// ---------------------------------------------------------------------------
__global__ __launch_bounds__(256) void k_dwconv2(
    const _Float16* __restrict__ t1, const float* __restrict__ dwW,
    const float* __restrict__ g2, const float* __restrict__ b2,
    _Float16* __restrict__ t2) {
  int idx = blockIdx.x * 256 + threadIdx.x;
  if (idx >= NTOK * DIN) return;
  int d = idx % DIN, tok = idx / DIN;
  int b = tok / L_, pos = tok % L_, h = pos / W_, w = pos % W_;
  float s = 0.f;
#pragma unroll
  for (int i = 0; i < 3; ++i)
#pragma unroll
    for (int j = 0; j < 3; ++j) {
      int hh = h + i - 1, ww = w + j - 1;
      if (hh >= 0 && hh < H_ && ww >= 0 && ww < W_)
        s += (float)t1[((size_t)b * L_ + hh * W_ + ww) * DIN + d] * dwW[d * 9 + i * 3 + j];
    }
  float v = s * g2[d] + b2[d];
  float ge = 0.5f * v * (1.f + erff(v * 0.70710678f));
  t2[idx] = (_Float16)ge;
}

// ---------------------------------------------------------------------------
// pw2 (192->96) + bn3 + residual + final LN -> out (f32)  -- async A staging
// ---------------------------------------------------------------------------
__global__ __launch_bounds__(128) void k_pw2_ln(
    const _Float16* __restrict__ t2, const _Float16* __restrict__ Wp6,
    const float* __restrict__ g3, const float* __restrict__ b3,
    const float* __restrict__ x2, const float* __restrict__ nw,
    const float* __restrict__ nb, float* __restrict__ out) {
  __shared__ __align__(16) _Float16 At[16][200];
  __shared__ float    Co[16][96];
  int tok0 = blockIdx.x * 16;
  int tid = threadIdx.x, lane = tid & 31, wave = tid >> 5;
  for (int c = tid; c < 16 * 24; c += 128) {
    int m = c / 24, q = c % 24;
    async_lds_b128(&At[m][q * 8], t2 + (size_t)(tok0 + m) * 192 + q * 8);
  }
  wait_async0();
  __syncthreads();
  const v16h* Bw = (const v16h*)Wp6;   // KT=6, NT=6
  for (int nt = wave; nt < 6; nt += 4) {
    v8f acc = {};
#pragma unroll
    for (int kt = 0; kt < 6; ++kt) {
      v16h a = fragA(&At[0][0] + kt * 32, 200, lane);
      v16h b = Bw[(size_t)(nt * 6 + kt) * 32 + lane];
      acc = wmma32f16(a, b, acc);
    }
    int n = nt * 16 + (lane & 15);
    float gg = g3[n], bb = b3[n];
    int hi = lane >> 4;
#pragma unroll
    for (int v = 0; v < 8; ++v) {
      int m = hi * 8 + v;
      int tok = tok0 + m;
      Co[m][n] = acc[v] * gg + bb + x2[(size_t)tok * 96 + n];
    }
  }
  __syncthreads();
  if (tid < 16) {
    float mean = 0.f, var = 0.f;
    for (int c = 0; c < 96; ++c) mean += Co[tid][c];
    mean *= (1.f / 96.f);
    for (int c = 0; c < 96; ++c) { float d = Co[tid][c] - mean; var += d * d; }
    var *= (1.f / 96.f);
    float r = rsqrtf(var + 1e-5f);
    int tok = tok0 + tid;
    for (int c = 0; c < 96; ++c)
      out[(size_t)tok * 96 + c] = (Co[tid][c] - mean) * r * nw[c] + nb[c];
  }
}

// ---------------------------------------------------------------------------
// Host launcher
// ---------------------------------------------------------------------------
extern "C" void kernel_launch(void* const* d_in, const int* in_sizes, int n_in,
                              void* d_out, int out_size, void* d_ws, size_t ws_size,
                              hipStream_t stream) {
  (void)in_sizes; (void)n_in; (void)out_size; (void)ws_size;
  const float* x_cat      = (const float*)d_in[0];
  const float* proj_W     = (const float*)d_in[1];
  const float* proj_b     = (const float*)d_in[2];
  const float* ln1_w      = (const float*)d_in[3];
  const float* ln1_b      = (const float*)d_in[4];
  const float* in_proj_W  = (const float*)d_in[5];
  const float* in_proj_b  = (const float*)d_in[6];
  const float* conv_W     = (const float*)d_in[7];
  const float* conv_b     = (const float*)d_in[8];
  const float* x_proj_W   = (const float*)d_in[9];
  const float* dt_W       = (const float*)d_in[10];
  const float* dt_b       = (const float*)d_in[11];
  const float* A_logs     = (const float*)d_in[12];
  const float* Ds         = (const float*)d_in[13];
  const float* out_norm_w = (const float*)d_in[14];
  const float* out_norm_b = (const float*)d_in[15];
  const float* out_proj_W = (const float*)d_in[16];
  const float* cb_pw1_W   = (const float*)d_in[17];
  const float* cb_bn1_g   = (const float*)d_in[18];
  const float* cb_bn1_b   = (const float*)d_in[19];
  const float* cb_dw_W    = (const float*)d_in[20];
  const float* cb_bn2_g   = (const float*)d_in[21];
  const float* cb_bn2_b   = (const float*)d_in[22];
  const float* cb_pw2_W   = (const float*)d_in[23];
  const float* cb_bn3_g   = (const float*)d_in[24];
  const float* cb_bn3_b   = (const float*)d_in[25];
  const float* norm_w     = (const float*)d_in[26];
  const float* norm_b     = (const float*)d_in[27];

  char* ws = (char*)d_ws;
  size_t off = 0;
  auto alloc = [&](size_t bytes) -> void* {
    void* p = ws + off;
    off += (bytes + 255) & ~(size_t)255;
    return p;
  };
  _Float16* Wp1 = (_Float16*)alloc((size_t)6 * 6 * 512 * 2);
  _Float16* Wp2 = (_Float16*)alloc((size_t)24 * 3 * 512 * 2);
  _Float16* Wp3 = (_Float16*)alloc((size_t)12 * 6 * 512 * 2);
  _Float16* Wp4 = (_Float16*)alloc((size_t)6 * 6 * 512 * 2);
  _Float16* Wp5 = (_Float16*)alloc((size_t)12 * 3 * 512 * 2);
  _Float16* Wp6 = (_Float16*)alloc((size_t)6 * 6 * 512 * 2);
  float*    x_res   = (float*)alloc((size_t)NTOK * 96 * 4);
  float*    z_g     = (float*)alloc((size_t)NTOK * 192 * 4);
  float*    xm_g    = (float*)alloc((size_t)NTOK * 192 * 4);
  _Float16* u_g     = (_Float16*)alloc((size_t)NTOK * 192 * 2);
  float*    delta_g = (float*)alloc((size_t)16 * L_ * 192 * 4);
  float*    Bs_g    = (float*)alloc((size_t)16 * L_ * 16 * 4);
  float*    Cs_g    = (float*)alloc((size_t)16 * L_ * 16 * 4);
  float*    ys_g    = (float*)alloc((size_t)4 * NTOK * 192 * 4);
  _Float16* yg_g    = (_Float16*)alloc((size_t)NTOK * 192 * 2);
  float*    x2_g    = (float*)alloc((size_t)NTOK * 96 * 4);
  _Float16* x2h_g   = (_Float16*)alloc((size_t)NTOK * 96 * 2);
  _Float16* t1_g    = (_Float16*)alloc((size_t)NTOK * 192 * 2);
  _Float16* t2_g    = (_Float16*)alloc((size_t)NTOK * 192 * 2);

  auto gp = [](int total) { return (total + 127) / 128; };
  k_prepack<<<gp(6 * 6 * 32),  128, 0, stream>>>(proj_W,     Wp1, 192, 96,  6, 6,  0);
  k_prepack<<<gp(24 * 3 * 32), 128, 0, stream>>>(in_proj_W,  Wp2, 96,  384, 3, 24, 0);
  k_prepack<<<gp(12 * 6 * 32), 128, 0, stream>>>(x_proj_W,   Wp3, 192, 192, 6, 12, 2);
  k_prepack<<<gp(6 * 6 * 32),  128, 0, stream>>>(out_proj_W, Wp4, 192, 96,  6, 6,  0);
  k_prepack<<<gp(12 * 3 * 32), 128, 0, stream>>>(cb_pw1_W,   Wp5, 96,  192, 3, 12, 1);
  k_prepack<<<gp(6 * 6 * 32),  128, 0, stream>>>(cb_pw2_W,   Wp6, 192, 96,  6, 6,  1);

  k_proj_ln_inproj<<<NTOK / 16, 128, 0, stream>>>(x_cat, proj_b, ln1_w, ln1_b,
                                                  in_proj_b, Wp1, Wp2,
                                                  x_res, xm_g, z_g);
  k_dwconv_silu<<<(NTOK * DIN + 255) / 256, 256, 0, stream>>>(xm_g, conv_W, conv_b, u_g);
  k_xproj<<<NTOK / 16, 128, 0, stream>>>(u_g, Wp3, dt_W, dt_b, delta_g, Bs_g, Cs_g);
  k_scan<<<96, 32, 0, stream>>>(delta_g, Bs_g, Cs_g, u_g, A_logs, Ds, ys_g);
  k_merge_ln_gate<<<NTOK, 192, 0, stream>>>(ys_g, z_g, out_norm_w, out_norm_b, yg_g);
  k_outproj<<<NTOK / 16, 128, 0, stream>>>(yg_g, Wp4, x_res, x2_g, x2h_g);
  k_pw1<<<NTOK / 16, 128, 0, stream>>>(x2h_g, Wp5, cb_bn1_g, cb_bn1_b, t1_g);
  k_dwconv2<<<(NTOK * DIN + 255) / 256, 256, 0, stream>>>(t1_g, cb_dw_W, cb_bn2_g, cb_bn2_b, t2_g);
  k_pw2_ln<<<NTOK / 16, 128, 0, stream>>>(t2_g, Wp6, cb_bn3_g, cb_bn3_b, x2_g,
                                          norm_w, norm_b, (float*)d_out);
}
